// Attention_3032246911698
// MI455X (gfx1250) — matrix operand order repaired and checked by reading, not compiled, not run on hardware
//
#include <hip/hip_runtime.h>
#include <hip/hip_bf16.h>

#define H      1024
#define LSEQ   2048
#define BATCH  16
#define KDIM   2048          // 2*H
#define NEGINF -10000000000.0f

typedef __attribute__((ext_vector_type(16))) __bf16 v16bf;
typedef __attribute__((ext_vector_type(8)))  float  v8f;

__device__ __forceinline__ unsigned short f2bf_bits(float f) {
    // round-to-nearest-even f32 -> bf16
    unsigned int u = __float_as_uint(f);
    unsigned int r = (u + 0x7FFFu + ((u >> 16) & 1u)) >> 16;
    return (unsigned short)r;
}

// Native CDNA5 transcendental; trailing v_nop satisfies the trans-op RAW
// hazard rule (1 independent op before result use) since the compiler cannot
// schedule around the asm contents.
__device__ __forceinline__ float fast_tanh(float x) {
    float y;
    asm("v_tanh_f32 %0, %1\n\tv_nop" : "=v"(y) : "v"(x));
    return y;
}

// ---------------------------------------------------------------------------
// Kernel A1: score_dec[b,h] = dec[b,:] . W_v[h,:] + attn_b[h]
// ---------------------------------------------------------------------------
__global__ __launch_bounds__(256) void score_dec_kernel(
    const float* __restrict__ v, const float* __restrict__ attn_w,
    const float* __restrict__ attn_b, float* __restrict__ sdec) {
    int idx = blockIdx.x * 256 + threadIdx.x;      // 0 .. 16383
    int b = idx >> 10;
    int h = idx & (H - 1);
    const float* dec = v + (size_t)b * H;
    const float* w   = attn_w + (size_t)h * (3 * H);   // W_v row h = attn_w[h, 0:H]
    float s = attn_b[h];
    for (int k = 0; k < H; ++k) s = fmaf(dec[k], w[k], s);
    sdec[idx] = s;
}

// ---------------------------------------------------------------------------
// Kernel A2: W_e^T -> bf16, pre-swizzled into WMMA B-fragment order.
// B[k][n] = attn_w[n, H + k].  Fragment f = nt*64 + kk covers K=[kk*32,+32),
// N=[nt*16,+16).  Contiguous k order (matches A staging): lane ln holds N = ln%16, K = kk*32 + (ln/16)*16 + j.
// Stored at wsB[f*512 + ln*16 + j]  (each lane's 16 bf16 contiguous = 32 B).
// ---------------------------------------------------------------------------
__global__ __launch_bounds__(256) void convertB_kernel(
    const float* __restrict__ attn_w, unsigned short* __restrict__ wsB) {
    int idx  = blockIdx.x * 256 + threadIdx.x;     // 0 .. 2M-1
    int f    = idx >> 9;
    int pos  = idx & 511;
    int ln   = pos >> 4;
    int j    = pos & 15;
    int nt   = f >> 6;
    int kk   = f & 63;
    int K = kk * 32 + (ln >> 4) * 16 + j;
    int N = nt * 16 + (ln & 15);
    wsB[idx] = f2bf_bits(attn_w[(size_t)N * (3 * H) + H + K]);
}

// ---------------------------------------------------------------------------
// Kernel B: fused score_enc GEMM (bf16 WMMA) + tanh + v_w dot + mask.
// Block = 32 rows of flattened (b*L + l)  (2 M-tiles), 256 threads = 8 waves.
// Wave w owns N-tiles nt = w*8 .. w*8+7 (covers all 64 N-tiles = H=1024).
// Each B-fragment load feeds TWO wmma (one per M-tile) -> halves L2 B traffic.
// ---------------------------------------------------------------------------
__global__ __launch_bounds__(256) void attn_score_kernel(
    const float* __restrict__ enc, const int* __restrict__ mask,
    const unsigned short* __restrict__ wsB, const float* __restrict__ sdec,
    const float* __restrict__ vw, float* __restrict__ raw) {

    __shared__ unsigned short ldsA[16 * 2 * 512]; // 32 KB: 16 K-steps x 2 M-tiles
    __shared__ float redw[8 * 32];                // per-wave row partials

    const int tid  = threadIdx.x;
    const int lane = tid & 31;
    const int wave = tid >> 5;
    const int row0 = blockIdx.x * 32;             // 32 rows, all in same batch b
    const int b    = row0 >> 11;                  // row0 / 2048

    v8f acc0[8] = {};
    v8f acc1[8] = {};
    const v16bf* Bv = reinterpret_cast<const v16bf*>(wsB);
    const float* encRow = enc + (size_t)row0 * KDIM;

    for (int chunk = 0; chunk < 4; ++chunk) {     // four 512-wide K chunks
        __syncthreads();                          // protect LDS reuse
        const int cbase = chunk * 512;
        // Vectorized staging: 4-aligned groups of 4 K-columns map to 4
        // consecutive j within one lane's fragment slice -> float4 global load,
        // 4x f32->bf16, one 8-byte LDS store.
        for (int i = 0; i < 16; ++i) {
            int q  = tid + (i << 8);              // vec4 index 0 .. 4095
            int e  = q << 2;                      // element index 0 .. 16383
            int r  = e >> 9;                      // row within 32-row tile
            int c  = e & 511;                     // K within chunk (mult of 4)
            int kk = c >> 5;                      // 0..15
            int k5 = c & 31;                      // 0,4,..,28
            int hh, j;
            hh = k5 >> 4; j = k5 & 15;
            int mt = r >> 4;
            int ln = hh * 16 + (r & 15);          // A: M = lane%16, K = 16*(lane/16) + j (contiguous, same as B)
            const float4 f4 = *reinterpret_cast<const float4*>(
                &encRow[(size_t)r * KDIM + cbase + c]);
            uint2 packed;
            packed.x = (unsigned int)f2bf_bits(f4.x) | ((unsigned int)f2bf_bits(f4.y) << 16);
            packed.y = (unsigned int)f2bf_bits(f4.z) | ((unsigned int)f2bf_bits(f4.w) << 16);
            *reinterpret_cast<uint2*>(&ldsA[((kk * 2 + mt) * 512) + ln * 16 + j]) = packed;
        }
        __syncthreads();

        for (int kk = 0; kk < 16; ++kk) {
            v16bf a0 = *reinterpret_cast<const v16bf*>(&ldsA[(kk * 2 + 0) * 512 + lane * 16]);
            v16bf a1 = *reinterpret_cast<const v16bf*>(&ldsA[(kk * 2 + 1) * 512 + lane * 16]);
            const int kkg = chunk * 16 + kk;
            #pragma unroll
            for (int i = 0; i < 8; ++i) {
                int nt = wave * 8 + i;
                v16bf bfrag = Bv[(size_t)(nt * 64 + kkg) * 32 + lane];
                acc0[i] = __builtin_amdgcn_wmma_f32_16x16x32_bf16(
                    false, a0, false, bfrag, (short)0, acc0[i], false, false);
                acc1[i] = __builtin_amdgcn_wmma_f32_16x16x32_bf16(
                    false, a1, false, bfrag, (short)0, acc1[i], false, false);
            }
        }
    }

    // Epilogue: partial[mt][j] = sum over this lane's 8 N-tiles of vw[n]*tanh(sd+score)
    // C/D layout: element j -> row M = mt*16 + j + 8*(lane/16), col N = Nbase + lane%16
    float p0[8], p1[8];
    #pragma unroll
    for (int j = 0; j < 8; ++j) { p0[j] = 0.0f; p1[j] = 0.0f; }

    #pragma unroll
    for (int i = 0; i < 8; ++i) {
        int n  = (wave * 8 + i) * 16 + (lane & 15);
        float sd = sdec[b * H + n];
        float w  = vw[n];
        #pragma unroll
        for (int j = 0; j < 8; ++j) {
            p0[j] += w * fast_tanh(sd + acc0[i][j]);
            p1[j] += w * fast_tanh(sd + acc1[i][j]);
        }
    }

    // deterministic butterfly reduce across the 16 lanes sharing each row
    const int mrow = (lane >> 4) * 8;
    #pragma unroll
    for (int j = 0; j < 8; ++j) {
        float a = p0[j];
        a += __shfl_xor(a, 1, 16);
        a += __shfl_xor(a, 2, 16);
        a += __shfl_xor(a, 4, 16);
        a += __shfl_xor(a, 8, 16);
        float c = p1[j];
        c += __shfl_xor(c, 1, 16);
        c += __shfl_xor(c, 2, 16);
        c += __shfl_xor(c, 4, 16);
        c += __shfl_xor(c, 8, 16);
        if ((lane & 15) == 0) {
            redw[wave * 32 + mrow + j]      = a;
            redw[wave * 32 + 16 + mrow + j] = c;
        }
    }
    __syncthreads();

    if (tid < 32) {
        float s = 0.0f;
        #pragma unroll
        for (int w = 0; w < 8; ++w) s += redw[w * 32 + tid];
        int grow = row0 + tid;
        raw[grow] = (mask[grow] == 0) ? NEGINF : s;
    }
}

// ---------------------------------------------------------------------------
// Kernel C: masked softmax over L per batch (mask already folded into raw)
// ---------------------------------------------------------------------------
__global__ __launch_bounds__(256) void softmax_kernel(
    const float* __restrict__ raw, float* __restrict__ out) {
    __shared__ float sm[256];
    const int b   = blockIdx.x;
    const int tid = threadIdx.x;
    const float* x = raw + (size_t)b * LSEQ;

    float m = -3.4e38f;
    for (int i = tid; i < LSEQ; i += 256) m = fmaxf(m, x[i]);
    sm[tid] = m; __syncthreads();
    for (int s = 128; s > 0; s >>= 1) {
        if (tid < s) sm[tid] = fmaxf(sm[tid], sm[tid + s]);
        __syncthreads();
    }
    m = sm[0]; __syncthreads();

    float sum = 0.0f;
    for (int i = tid; i < LSEQ; i += 256) sum += __expf(x[i] - m);
    sm[tid] = sum; __syncthreads();
    for (int s = 128; s > 0; s >>= 1) {
        if (tid < s) sm[tid] += sm[tid + s];
        __syncthreads();
    }
    float inv = 1.0f / sm[0];

    for (int i = tid; i < LSEQ; i += 256)
        out[(size_t)b * LSEQ + i] = __expf(x[i] - m) * inv;
}

// ---------------------------------------------------------------------------
extern "C" void kernel_launch(void* const* d_in, const int* in_sizes, int n_in,
                              void* d_out, int out_size, void* d_ws, size_t ws_size,
                              hipStream_t stream) {
    const float* enc    = (const float*)d_in[0];   // (16, 2048, 2048) f32
    const int*   mask   = (const int*)  d_in[1];   // (16, 2048) i32
    const float* v      = (const float*)d_in[2];   // (1, 16, 1024) f32
    const float* attn_w = (const float*)d_in[3];   // (1024, 3072) f32
    const float* attn_b = (const float*)d_in[4];   // (1024,) f32
    const float* v_w    = (const float*)d_in[5];   // (1, 1024) f32
    float* out = (float*)d_out;                    // (16, 2048) f32

    // workspace layout
    unsigned short* wsB = (unsigned short*)d_ws;                           // 4 MiB
    float* sdec = (float*)((char*)d_ws + (size_t)4 * 1024 * 1024);         // 64 KiB
    float* raw  = sdec + BATCH * H;                                        // 128 KiB

    score_dec_kernel<<<(BATCH * H) / 256, 256, 0, stream>>>(v, attn_w, attn_b, sdec);
    convertB_kernel<<<(KDIM * H) / 256, 256, 0, stream>>>(attn_w, wsB);
    attn_score_kernel<<<(BATCH * LSEQ) / 32, 256, 0, stream>>>(enc, mask, wsB, sdec, v_w, raw);
    softmax_kernel<<<BATCH, 256, 0, stream>>>(raw, out);
}
